// CNN_3D_mask_6828998001192
// MI455X (gfx1250) — compile-verified, matrix operand order checked by reading, not run on hardware
//
#include <hip/hip_runtime.h>
#include <hip/hip_bf16.h>

typedef __bf16 bf16;
typedef __attribute__((ext_vector_type(4)))  __bf16 v4bf;
typedef __attribute__((ext_vector_type(8)))  __bf16 v8bf;
typedef __attribute__((ext_vector_type(16))) __bf16 v16bf;
typedef __attribute__((ext_vector_type(8)))  float  v8f;

// ---------------- workspace layout (bytes) ----------------
static constexpr size_t MB_ = 1024ull * 1024ull;
static constexpr size_t OFF_E     = 0;          // 800*225*4 fp32 embeddings
static constexpr size_t OFF_STATS = 1 * MB_;    // 9 stages * 256 floats: [sum|sumsq|scale|shift]
static constexpr size_t OFF_Z     = 2 * MB_;    // 800*25*225 fp32 = 18 MB
static constexpr size_t OFF_PRE   = 22 * MB_;   // 18 MB (pre-BN 2D-conv out)
static constexpr size_t OFF_BIGA  = 42 * MB_;   // 288 MB bf16 (h1, later h3)
static constexpr size_t OFF_BIGB  = 330 * MB_;  // 288 MB bf16 (h2, later h4)

// ---------------- embedding gather ----------------
__global__ void __launch_bounds__(256)
embed_kernel(const int* __restrict__ x, const float* __restrict__ emb, float* __restrict__ eF) {
  int bi = blockIdx.x, tid = threadIdx.x;
  if (tid < 225) eF[(size_t)bi * 225 + tid] = emb[(size_t)x[bi] * 225 + tid];
}

// ---------------- conv3a: Cin=1, Cout=4, K=27 -> scalar (masked input) ----------------
__global__ void __launch_bounds__(256)
conv3a_kernel(const float* __restrict__ eF, const float* __restrict__ w3a,
              bf16* __restrict__ h1, float* __restrict__ stats) {
  int bi = blockIdx.x, zo = blockIdx.y, tid = threadIdx.x;
  int b = bi / 200, i = bi - b * 200;
  __shared__ float s[3 * 289];
  __shared__ float wA[108];
  __shared__ float ssum[4], ssq[4];
  if (tid < 4) { ssum[tid] = 0.f; ssq[tid] = 0.f; }
  if (tid < 108) wA[tid] = w3a[tid];
  for (int idx = tid; idx < 3 * 289; idx += 256) {
    int zz = idx / 289, q = idx - zz * 289, yy = q / 17, xx = q - yy * 17;
    int j = zo - 1 + zz, sy = yy - 1, sx = xx - 1;
    float v = 0.f;
    if (j >= 0 && j < 200 && j <= i && sy >= 0 && sy < 15 && sx >= 0 && sx < 15)
      v = eF[((size_t)b * 200 + j) * 225 + sy * 15 + sx];   // causal mask j<=i
    s[idx] = v;
  }
  __syncthreads();
  for (int idx = tid; idx < 4 * 225; idx += 256) {
    int c = idx / 225, p = idx - c * 225, y = p / 15, x = p - y * 15;
    float acc = 0.f;
    for (int dz = 0; dz < 3; dz++)
      for (int dy = 0; dy < 3; dy++)
        for (int dx = 0; dx < 3; dx++)
          acc += s[dz * 289 + (y + dy) * 17 + (x + dx)] * wA[c * 27 + dz * 9 + dy * 3 + dx];
    h1[(((size_t)bi * 4 + c) * 200 + zo) * 225 + p] = (bf16)acc;
    atomicAdd(&ssum[c], acc);
    atomicAdd(&ssq[c], acc * acc);
  }
  __syncthreads();
  if (tid < 4) { atomicAdd(&stats[tid], ssum[tid]); atomicAdd(&stats[64 + tid], ssq[tid]); }
}

// ---------------- BN finalize: mean/var -> scale/shift ----------------
__global__ void bn_finalize_kernel(float* __restrict__ stats, const float* __restrict__ g,
                                   const float* __restrict__ b, int nch, float count) {
  int c = threadIdx.x;
  if (c < nch) {
    float m   = stats[c] / count;
    float var = stats[64 + c] / count - m * m;
    float inv = rsqrtf(var + 1e-5f);
    float sc  = g[c] * inv;
    stats[128 + c] = sc;
    stats[192 + c] = b[c] - m * sc;
  }
}

// ---------------- generic strided 3D conv via implicit-GEMM WMMA ----------------
// K-order: k = tap*CIN + cin (channel-innermost LDS slab -> contiguous wide gathers).
// Padded taps point at a zeroed slab slice -> all LDS loads unconditional.
template <int CIN, int COUT, int DIN, int DOUT>
__global__ void __launch_bounds__(128)
conv3_wmma_kernel(const bf16* __restrict__ in, const float* __restrict__ prevStats,
                  const float* __restrict__ w, bf16* __restrict__ out, float* __restrict__ stats) {
  constexpr int TAPP = (CIN == 4) ? 32 : 28;   // taps padded so KP % 32 == 0
  constexpr int KP   = TAPP * CIN;             // 128 / 224
  constexpr int NCH  = KP / 32;                // 4 / 7
  constexpr int ZOFF = 3 * 289 * CIN;          // zero slab slice
  constexpr int STOT = 4 * 289 * CIN;

  const int bi = blockIdx.x, d = blockIdx.y;
  const int tid = threadIdx.x, lane = tid & 31, wv = tid >> 5;

  __shared__ alignas(16) bf16 sIn[STOT];
  __shared__ alignas(16) bf16 sW[16 * KP];
  __shared__ float ssum[16], ssq[16];

  if (tid < 16) { ssum[tid] = 0.f; ssq[tid] = 0.f; }
  const float* scl = prevStats + 128;
  const float* sft = prevStats + 192;

  // stage halo slab: addr = ((zz*17+yy)*17+xx)*CIN + cin ; zz==3 -> zero slice
  for (int idx = tid; idx < STOT; idx += 128) {
    int cin = idx % CIN;
    int pos = idx / CIN;
    int xx = pos % 17, r1 = pos / 17;
    int yy = r1 % 17, zz = r1 / 17;
    int zi = 2 * d + zz - 1, sy = yy - 1, sx = xx - 1;
    float v = 0.f;
    if (zz < 3 && zi >= 0 && zi < DIN && sy >= 0 && sy < 15 && sx >= 0 && sx < 15) {
      float f = (float)in[(((size_t)bi * CIN + cin) * DIN + zi) * 225 + sy * 15 + sx];
      v = fmaxf(f * scl[cin] + sft[cin], 0.f);   // fused BN+ReLU of previous layer
    }
    sIn[idx] = (bf16)v;
  }
  // stage padded weights (A matrix), k = tap*CIN + cin
  for (int idx = tid; idx < 16 * KP; idx += 128) {
    int m = idx / KP, k = idx - m * KP;
    int tap = k / CIN, cin = k % CIN;
    float v = 0.f;
    if (m < COUT && tap < 27) v = w[(m * CIN + cin) * 27 + tap];
    sW[idx] = (bf16)v;
  }
  __syncthreads();

  const int khalf = (lane >= 16) ? 8 : 0;
  const int nn = lane & 15, mrow = lane & 15, mbase = (lane >= 16) ? 8 : 0;

  auto tapbase = [&](int tap) -> int {
    if (tap >= 27) return ZOFF;
    int dz = tap / 9, r = tap - dz * 9, dy = r / 3, dx = r - dy * 3;
    return ((dz * 17 + dy) * 17 + dx) * CIN;
  };
  int tb[NCH][4];
#pragma unroll
  for (int kc = 0; kc < NCH; ++kc) {
    if constexpr (CIN == 8) {
      int t0 = kc * 4 + (khalf >> 3);
      tb[kc][0] = tapbase(t0);
      tb[kc][1] = tapbase(t0 + 2);
    } else {
      int t0 = kc * 8 + (khalf >> 2);
      tb[kc][0] = tapbase(t0);
      tb[kc][1] = tapbase(t0 + 1);
      tb[kc][2] = tapbase(t0 + 4);
      tb[kc][3] = tapbase(t0 + 5);
    }
  }

  float rsum[8], rsq[8];
#pragma unroll
  for (int r = 0; r < 8; ++r) { rsum[r] = 0.f; rsq[r] = 0.f; }

  for (int t = wv; t < 15; t += 4) {           // uniform per wave -> EXEC all-ones at WMMA
    int n = t * 16 + nn;
    int ncl = (n < 225) ? n : 224;
    int y = ncl / 15, x = ncl - y * 15;
    int yxs = (y * 17 + x) * CIN;
    v8f acc = {};
#pragma unroll
    for (int kc = 0; kc < NCH; ++kc) {
      const int arow = mrow * KP + kc * 32 + khalf;
      v8bf a0 = *(const v8bf*)(sW + arow);
      v8bf a1 = *(const v8bf*)(sW + arow + 16);
      v16bf a = __builtin_shufflevector(a0, a1, 0,1,2,3,4,5,6,7,8,9,10,11,12,13,14,15);
      v16bf b;
      if constexpr (CIN == 8) {
        v8bf g0 = *(const v8bf*)(sIn + tb[kc][0] + yxs);
        v8bf g1 = *(const v8bf*)(sIn + tb[kc][1] + yxs);
        b = __builtin_shufflevector(g0, g1, 0,1,2,3,4,5,6,7,8,9,10,11,12,13,14,15);
      } else {
        v4bf g0 = *(const v4bf*)(sIn + tb[kc][0] + yxs);
        v4bf g1 = *(const v4bf*)(sIn + tb[kc][1] + yxs);
        v4bf g2 = *(const v4bf*)(sIn + tb[kc][2] + yxs);
        v4bf g3 = *(const v4bf*)(sIn + tb[kc][3] + yxs);
        v8bf b01 = __builtin_shufflevector(g0, g1, 0,1,2,3,4,5,6,7);
        v8bf b23 = __builtin_shufflevector(g2, g3, 0,1,2,3,4,5,6,7);
        b = __builtin_shufflevector(b01, b23, 0,1,2,3,4,5,6,7,8,9,10,11,12,13,14,15);
      }
      acc = __builtin_amdgcn_wmma_f32_16x16x32_bf16(false, a, false, b, (short)0, acc, false, false);
    }
    bool nval = (n < 225);
#pragma unroll
    for (int r = 0; r < 8; ++r) {
      int m = r + mbase;                        // D layout: M = r + (lane>=16 ? 8 : 0)
      bool val = nval && (m < COUT);
      float v = acc[r];
      if (val) out[(((size_t)bi * COUT + m) * DOUT + d) * 225 + n] = (bf16)v;
      float vm = val ? v : 0.f;
      rsum[r] += vm;
      rsq[r] += vm * vm;
    }
  }
#pragma unroll
  for (int r = 0; r < 8; ++r) {
    int m = r + mbase;
    if (m < COUT) { atomicAdd(&ssum[m], rsum[r]); atomicAdd(&ssq[m], rsq[r]); }
  }
  __syncthreads();
  if (tid < COUT) { atomicAdd(&stats[tid], ssum[tid]); atomicAdd(&stats[64 + tid], ssq[tid]); }
}

// ---------------- conv3d: Cin=4, Cout=1 -> scalar ----------------
__global__ void __launch_bounds__(256)
conv3d_kernel(const bf16* __restrict__ h3, const float* __restrict__ statsC,
              const float* __restrict__ w3d, bf16* __restrict__ h4, float* __restrict__ stats) {
  const int bi = blockIdx.x, d = blockIdx.y, tid = threadIdx.x;
  __shared__ bf16 s[12 * 289];
  __shared__ float wD[108];
  __shared__ float ssum1[1], ssq1[1];
  if (tid == 0) { ssum1[0] = 0.f; ssq1[0] = 0.f; }
  if (tid < 108) wD[tid] = w3d[tid];
  const float* scl = statsC + 128;
  const float* sft = statsC + 192;
  for (int idx = tid; idx < 12 * 289; idx += 256) {
    int zz = idx / (4 * 289), rem = idx - zz * 4 * 289;
    int cin = rem / 289, q = rem - cin * 289;
    int yy = q / 17, xx = q - yy * 17;
    int zi = 2 * d + zz - 1, sy = yy - 1, sx = xx - 1;
    float v = 0.f;
    if (zi >= 0 && zi < 50 && sy >= 0 && sy < 15 && sx >= 0 && sx < 15) {
      float f = (float)h3[(((size_t)bi * 4 + cin) * 50 + zi) * 225 + sy * 15 + sx];
      v = fmaxf(f * scl[cin] + sft[cin], 0.f);
    }
    s[idx] = (bf16)v;
  }
  __syncthreads();
  if (tid < 225) {
    int y = tid / 15, x = tid - y * 15;
    float acc = 0.f;
    for (int cin = 0; cin < 4; cin++)
      for (int dz = 0; dz < 3; dz++)
        for (int dy = 0; dy < 3; dy++)
          for (int dx = 0; dx < 3; dx++)
            acc += (float)s[(dz * 4 + cin) * 289 + (y + dy) * 17 + (x + dx)] *
                   wD[cin * 27 + dz * 9 + dy * 3 + dx];
    h4[((size_t)bi * 25 + d) * 225 + tid] = (bf16)acc;
    atomicAdd(&ssum1[0], acc);
    atomicAdd(&ssq1[0], acc * acc);
  }
  __syncthreads();
  if (tid == 0) { atomicAdd(&stats[0], ssum1[0]); atomicAdd(&stats[64], ssq1[0]); }
}

// ---------------- z init: z = relu(bn_d(h4)) ----------------
__global__ void __launch_bounds__(256)
zinit_kernel(const bf16* __restrict__ h4, const float* __restrict__ statsD,
             float* __restrict__ z, int n) {
  int i = blockIdx.x * 256 + threadIdx.x;
  if (i < n) {
    float sc = statsD[128], sh = statsD[192];
    z[i] = fmaxf((float)h4[i] * sc + sh, 0.f);
  }
}

// ---------------- residual 2D conv via implicit-GEMM WMMA ----------------
// cin padded 25->32; k = tap*32 + cin; KP = 9*32 = 288 (one tap per K chunk).
__global__ void __launch_bounds__(128)
conv2_wmma_kernel(const float* __restrict__ z, const float* __restrict__ w,
                  float* __restrict__ pre, float* __restrict__ stats) {
  constexpr int KP = 288;
  const int bi = blockIdx.x;
  const int tid = threadIdx.x, lane = tid & 31, wv = tid >> 5;
  __shared__ alignas(16) bf16 sIn[289 * 32];
  __shared__ alignas(16) bf16 sW[32 * KP];
  __shared__ float ssum[32], ssq[32];
  if (tid < 32) { ssum[tid] = 0.f; ssq[tid] = 0.f; }
  for (int idx = tid; idx < 289 * 32; idx += 128) {
    int cin = idx & 31, pos = idx >> 5;
    int xx = pos % 17, yy = pos / 17;
    int sy = yy - 1, sx = xx - 1;
    float v = 0.f;
    if (cin < 25 && sy >= 0 && sy < 15 && sx >= 0 && sx < 15)
      v = z[((size_t)bi * 25 + cin) * 225 + sy * 15 + sx];
    sIn[idx] = (bf16)v;
  }
  for (int idx = tid; idx < 32 * KP; idx += 128) {
    int m = idx / KP, k = idx - m * KP;
    int tap = k >> 5, cin = k & 31;
    float v = 0.f;
    if (m < 25 && cin < 25) v = w[(m * 25 + cin) * 9 + tap];
    sW[idx] = (bf16)v;
  }
  __syncthreads();
  const int khalf = (lane >= 16) ? 8 : 0;
  const int nn = lane & 15, mrow = lane & 15, mbase = (lane >= 16) ? 8 : 0;
  int tb[9];
#pragma unroll
  for (int tap = 0; tap < 9; ++tap) {
    int dy = tap / 3, dx = tap - dy * 3;
    tb[tap] = (dy * 17 + dx) * 32;
  }
  float rsum[2][8], rsq[2][8];
#pragma unroll
  for (int mt = 0; mt < 2; ++mt)
#pragma unroll
    for (int r = 0; r < 8; ++r) { rsum[mt][r] = 0.f; rsq[mt][r] = 0.f; }

  for (int task = wv; task < 30; task += 4) {  // 15 N-tiles x 2 M-tiles
    int mt = task & 1, t = task >> 1;
    int n = t * 16 + nn;
    int ncl = (n < 225) ? n : 224;
    int y = ncl / 15, x = ncl - y * 15;
    int yxs = (y * 17 + x) * 32;
    v8f acc = {};
#pragma unroll
    for (int kc = 0; kc < 9; ++kc) {
      int arow = (mt * 16 + mrow) * KP + kc * 32 + khalf;
      v8bf a0 = *(const v8bf*)(sW + arow);
      v8bf a1 = *(const v8bf*)(sW + arow + 16);
      v16bf a = __builtin_shufflevector(a0, a1, 0,1,2,3,4,5,6,7,8,9,10,11,12,13,14,15);
      int gb = tb[kc] + yxs + khalf;
      v8bf g0 = *(const v8bf*)(sIn + gb);
      v8bf g1 = *(const v8bf*)(sIn + gb + 16);
      v16bf b = __builtin_shufflevector(g0, g1, 0,1,2,3,4,5,6,7,8,9,10,11,12,13,14,15);
      acc = __builtin_amdgcn_wmma_f32_16x16x32_bf16(false, a, false, b, (short)0, acc, false, false);
    }
    bool nval = (n < 225);
#pragma unroll
    for (int r = 0; r < 8; ++r) {
      int m = mt * 16 + r + mbase;
      bool val = nval && (m < 25);
      float v = acc[r];
      if (val) pre[((size_t)bi * 25 + m) * 225 + n] = v;
      float vm = val ? v : 0.f;
      rsum[mt][r] += vm;
      rsq[mt][r] += vm * vm;
    }
  }
#pragma unroll
  for (int mt = 0; mt < 2; ++mt)
#pragma unroll
    for (int r = 0; r < 8; ++r) {
      int m = mt * 16 + r + mbase;
      if (m < 25) { atomicAdd(&ssum[m], rsum[mt][r]); atomicAdd(&ssq[m], rsq[mt][r]); }
    }
  __syncthreads();
  if (tid < 25) { atomicAdd(&stats[tid], ssum[tid]); atomicAdd(&stats[64 + tid], ssq[tid]); }
}

// ---------------- z += relu(bn(pre)) ----------------
__global__ void __launch_bounds__(256)
apply2_kernel(float* __restrict__ z, const float* __restrict__ pre,
              const float* __restrict__ st, int n) {
  int i = blockIdx.x * 256 + threadIdx.x;
  if (i < n) {
    int ch = (i / 225) % 25;
    z[i] += fmaxf(pre[i] * st[128 + ch] + st[192 + ch], 0.f);
  }
}

// ---------------- head: pool + concat + fc1(tanh) + fc2(sigmoid) ----------------
__global__ void __launch_bounds__(256)
head_kernel(const float* __restrict__ z, const float* __restrict__ eF,
            const float* __restrict__ fc1w, const float* __restrict__ fc1b,
            const float* __restrict__ fc2w, const float* __restrict__ fc2b,
            float* __restrict__ out) {
  int bi = blockIdx.x, tid = threadIdx.x;
  __shared__ float feat[450];
  __shared__ float hfc[200];
  __shared__ float red[256];
  if (tid < 225) {
    float s = 0.f;
    for (int c = 0; c < 25; c++) s += z[((size_t)bi * 25 + c) * 225 + tid];
    feat[tid] = s * (1.f / 25.f);
    feat[225 + tid] = eF[(size_t)bi * 225 + tid];
  }
  __syncthreads();
  if (tid < 200) {
    float a = fc1b[tid];
    for (int k = 0; k < 450; k++) a += feat[k] * fc1w[tid * 450 + k];
    hfc[tid] = tanhf(a);
  }
  __syncthreads();
  red[tid] = (tid < 200) ? hfc[tid] * fc2w[tid] : 0.f;
  __syncthreads();
  for (int s = 128; s > 0; s >>= 1) {
    if (tid < s) red[tid] += red[tid + s];
    __syncthreads();
  }
  if (tid == 0) out[bi] = 1.f / (1.f + expf(-(red[0] + fc2b[0])));
}

// ---------------- host driver ----------------
extern "C" void kernel_launch(void* const* d_in, const int* in_sizes, int n_in,
                              void* d_out, int out_size, void* d_ws, size_t ws_size,
                              hipStream_t stream) {
  (void)in_sizes; (void)n_in; (void)out_size; (void)ws_size;
  const int*   x    = (const int*)d_in[0];
  const float* emb  = (const float*)d_in[1];
  const float* w3a  = (const float*)d_in[2];
  const float* g3a  = (const float*)d_in[3];
  const float* b3a  = (const float*)d_in[4];
  const float* w3b  = (const float*)d_in[5];
  const float* g3b  = (const float*)d_in[6];
  const float* b3b  = (const float*)d_in[7];
  const float* w3c  = (const float*)d_in[8];
  const float* g3c  = (const float*)d_in[9];
  const float* b3c  = (const float*)d_in[10];
  const float* w3d  = (const float*)d_in[11];
  const float* g3d  = (const float*)d_in[12];
  const float* b3d  = (const float*)d_in[13];
  const float* w2   = (const float*)d_in[14];
  const float* g2   = (const float*)d_in[15];
  const float* b2   = (const float*)d_in[16];
  const float* fc1w = (const float*)d_in[17];
  const float* fc1b = (const float*)d_in[18];
  const float* fc2w = (const float*)d_in[19];
  const float* fc2b = (const float*)d_in[20];

  char*  ws    = (char*)d_ws;
  float* eF    = (float*)(ws + OFF_E);
  float* stats = (float*)(ws + OFF_STATS);
  float* z     = (float*)(ws + OFF_Z);
  float* pre   = (float*)(ws + OFF_PRE);
  bf16*  bigA  = (bf16*)(ws + OFF_BIGA);
  bf16*  bigB  = (bf16*)(ws + OFF_BIGB);
  float* out   = (float*)d_out;

  hipMemsetAsync(stats, 0, 9 * 256 * sizeof(float), stream);

  embed_kernel<<<800, 256, 0, stream>>>(x, emb, eF);

  conv3a_kernel<<<dim3(800, 200), 256, 0, stream>>>(eF, w3a, bigA, stats + 0 * 256);
  bn_finalize_kernel<<<1, 64, 0, stream>>>(stats + 0 * 256, g3a, b3a, 4, 36000000.f);

  conv3_wmma_kernel<4, 8, 200, 100>
      <<<dim3(800, 100), 128, 0, stream>>>(bigA, stats + 0 * 256, w3b, bigB, stats + 1 * 256);
  bn_finalize_kernel<<<1, 64, 0, stream>>>(stats + 1 * 256, g3b, b3b, 8, 18000000.f);

  conv3_wmma_kernel<8, 4, 100, 50>
      <<<dim3(800, 50), 128, 0, stream>>>(bigB, stats + 1 * 256, w3c, bigA, stats + 2 * 256);
  bn_finalize_kernel<<<1, 64, 0, stream>>>(stats + 2 * 256, g3c, b3c, 4, 9000000.f);

  conv3d_kernel<<<dim3(800, 25), 256, 0, stream>>>(bigA, stats + 2 * 256, w3d, bigB, stats + 3 * 256);
  bn_finalize_kernel<<<1, 64, 0, stream>>>(stats + 3 * 256, g3d, b3d, 1, 4500000.f);

  zinit_kernel<<<(4500000 + 255) / 256, 256, 0, stream>>>(bigB, stats + 3 * 256, z, 4500000);

  for (int i = 0; i < 5; i++) {
    conv2_wmma_kernel<<<800, 128, 0, stream>>>(z, w2 + (size_t)i * 25 * 25 * 9, pre,
                                               stats + (4 + i) * 256);
    bn_finalize_kernel<<<1, 64, 0, stream>>>(stats + (4 + i) * 256, g2 + i * 25, b2 + i * 25,
                                             25, 180000.f);
    apply2_kernel<<<(4500000 + 255) / 256, 256, 0, stream>>>(z, pre, stats + (4 + i) * 256, 4500000);
  }

  head_kernel<<<800, 256, 0, stream>>>(z, eF, fc1w, fc1b, fc2w, fc2b, out);
}
